// MultiHeadAttention_Parallel_16518444221282
// MI455X (gfx1250) — matrix operand order repaired and checked by reading, not compiled, not run on hardware
//
#include <hip/hip_runtime.h>

#define BB 2
#define TT 2048
#define CC 1024
#define NHEAD 16
#define DH 64

typedef unsigned short ushort16_t;
typedef __attribute__((ext_vector_type(16))) __bf16 v16bf;
typedef __attribute__((ext_vector_type(8)))  __bf16 v8bf;
typedef __attribute__((ext_vector_type(8)))  float  v8f;

union ABFrag {
    v16bf v;
    struct { v8bf lo, hi; } h;
};

__device__ __forceinline__ ushort16_t f2bf(float f) {
    unsigned int u = __float_as_uint(f);
    u += 0x7FFFu + ((u >> 16) & 1u);          // round to nearest even
    return (ushort16_t)(u >> 16);
}

// gfx1250 async copy: 16B per lane, global -> LDS, tracked by ASYNCcnt.
// GVS mode: 64-bit SGPR base + 32-bit per-lane byte offset.
__device__ __forceinline__ void async_b128(unsigned lds_addr, unsigned goff,
                                           unsigned long long base) {
    asm volatile("global_load_async_to_lds_b128 %0, %1, %2"
                 :: "v"(lds_addr), "v"(goff), "s"(base) : "memory");
}

// ---------------------------------------------------------------- converts
__global__ void k_f32_to_bf16(const float* __restrict__ in,
                              ushort16_t* __restrict__ out, int n) {
    int i = blockIdx.x * blockDim.x + threadIdx.x;
    int stride = gridDim.x * blockDim.x;
    for (; i < n; i += stride) out[i] = f2bf(in[i]);
}

// Wt[n][k] = W[k][n], bf16.  Coalesced writes.
__global__ void k_w_transpose_bf16(const float* __restrict__ W,
                                   ushort16_t* __restrict__ Wt, int K, int N) {
    int idx = blockIdx.x * blockDim.x + threadIdx.x;
    if (idx >= K * N) return;
    int n = idx / K, k = idx % K;
    Wt[idx] = f2bf(W[(size_t)k * N + n]);
}

// Vt[b][h][d][t] = V[b*T+t][h*DH+d]  (bf16 -> bf16)
__global__ void k_v_transpose(const ushort16_t* __restrict__ V,
                              ushort16_t* __restrict__ Vt) {
    int idx = blockIdx.x * blockDim.x + threadIdx.x;
    if (idx >= BB * NHEAD * DH * TT) return;
    int t = idx % TT;
    int d = (idx / TT) % DH;
    int h = (idx / (TT * DH)) % NHEAD;
    int b = idx / (TT * DH * NHEAD);
    Vt[idx] = V[((size_t)(b * TT + t)) * CC + h * DH + d];
}

// ---------------------------------------------------------------- GEMM
// C[M,N] = A[M,K](bf16) * Bt[N,K](bf16, pre-transposed), f32 accum.
// Block = 8 waves -> 256x64 tile (32x64 per wave, 8 WMMAs per K-step).
// A/B K-panels double-buffered in LDS via global_load_async_to_lds_b128,
// synced with s_wait_asynccnt + barriers.  block(32,8), grid(N/64, M/256).
__global__ void k_gemm_bf16(const ushort16_t* __restrict__ A,
                            const ushort16_t* __restrict__ Bt,
                            float* __restrict__ Cf,
                            ushort16_t* __restrict__ Cb,
                            const float* __restrict__ bias,
                            int M, int N, int K) {
    __shared__ __align__(16) ushort16_t sA[2][256][40];  // 256 rows x 32 k (+pad)
    __shared__ __align__(16) ushort16_t sB[2][64][40];   //  64 cols x 32 k (+pad)

    const int lane  = threadIdx.x;
    const int wid   = threadIdx.y;
    const int tid   = wid * 32 + lane;
    const int m0blk = blockIdx.y * 256;
    const int n0    = blockIdx.x * 64;
    const int lhalf = (lane < 16) ? 0 : 1;
    const int lm    = lane & 15;

    const unsigned long long baseA = (unsigned long long)A;
    const unsigned long long baseB = (unsigned long long)Bt;
    const int rowBytes = K * 2;

    // stage one 32-wide K panel of A (256 rows) and B (64 rows) into LDS[buf]
    auto stage = [&](int buf, int kb) {
        // B: 64 rows x 64B = 256 x 16B chunks, one per thread
        {
            int r = tid >> 2, c = (tid & 3) * 16;
            unsigned lds = (unsigned)(size_t)&sB[buf][r][0] + (unsigned)c;
            unsigned go  = (unsigned)((n0 + r) * rowBytes + kb * 2 + c);
            async_b128(lds, go, baseB);
        }
        // A: 256 rows x 64B = 1024 x 16B chunks, four per thread
        #pragma unroll
        for (int i = 0; i < 4; ++i) {
            int chunk = tid + i * 256;
            int r = chunk >> 2, c = (chunk & 3) * 16;
            unsigned lds = (unsigned)(size_t)&sA[buf][r][0] + (unsigned)c;
            unsigned go  = (unsigned)((m0blk + r) * rowBytes + kb * 2 + c);
            async_b128(lds, go, baseA);
        }
    };

    v8f acc[2][4];
    #pragma unroll
    for (int r = 0; r < 2; ++r)
        #pragma unroll
        for (int t = 0; t < 4; ++t)
            acc[r][t] = (v8f){0.f,0.f,0.f,0.f,0.f,0.f,0.f,0.f};

    stage(0, 0);
    int it = 0;
    for (int kb = 0; kb < K; kb += 32, ++it) {
        const int cur = it & 1;
        const bool more = (kb + 32) < K;
        if (more) {
            stage(cur ^ 1, kb + 32);                       // prefetch next panel
            asm volatile("s_wait_asynccnt 5" ::: "memory"); // current panel done
        } else {
            asm volatile("s_wait_asynccnt 0" ::: "memory");
        }
        __syncthreads();                                   // panel visible to all

        // batch ALL fragment loads, then back-to-back WMMAs
        ABFrag a[2];
        #pragma unroll
        for (int r = 0; r < 2; ++r) {
            const ushort16_t* pa = &sA[cur][wid * 32 + r * 16 + lm][lhalf * 8];
            a[r].h.lo = *(const v8bf*)pa;          // K 0..7  (lanes>=16: 8..15)
            a[r].h.hi = *(const v8bf*)(pa + 16);   // K 16..23 (lanes>=16: 24..31)
        }
        ABFrag bf[4];
        #pragma unroll
        for (int t = 0; t < 4; ++t) {
            const ushort16_t* pb = &sB[cur][16 * t + lm][lhalf * 8];
            bf[t].h.lo = *(const v8bf*)pb;        // K 0..7  (lanes>=16: 8..15)
            bf[t].h.hi = *(const v8bf*)(pb + 16); // K 16..23 (lanes>=16: 24..31)
        }
        __builtin_amdgcn_sched_barrier(0);        // keep loads as one clause
        #pragma unroll
        for (int r = 0; r < 2; ++r)
            #pragma unroll
            for (int t = 0; t < 4; ++t)
                acc[r][t] = __builtin_amdgcn_wmma_f32_16x16x32_bf16(
                    false, a[r].v, false, bf[t].v, (short)0, acc[r][t], false, false);

        __syncthreads();            // everyone done reading before buf is reused
    }

    #pragma unroll
    for (int r = 0; r < 2; ++r)
        #pragma unroll
        for (int t = 0; t < 4; ++t) {
            int col = n0 + 16 * t + lm;
            float bv = bias ? bias[col] : 0.f;
            #pragma unroll
            for (int v = 0; v < 8; ++v) {
                int row = m0blk + wid * 32 + r * 16 + v + lhalf * 8;
                float val = acc[r][t][v] + bv;
                if (Cf) Cf[(size_t)row * N + col] = val;
                else    Cb[(size_t)row * N + col] = f2bf(val);
            }
        }
}

// ---------------------------------------------------------------- attention
// Block = 8 waves = 8 consecutive 16-query tiles of the SAME (b,h).
// K/V tiles for each 32-key block are staged once per block into
// double-buffered LDS via async loads (8x traffic reduction), waves skip
// compute uniformly past their causal limit.  block(32,8), grid(512).
__global__ void k_attn(const ushort16_t* __restrict__ Q,
                       const ushort16_t* __restrict__ Kc,
                       const ushort16_t* __restrict__ Vt,
                       ushort16_t* __restrict__ O) {
    __shared__ float sS[8][16][36];
    __shared__ __align__(16) ushort16_t sP[8][16][40];
    __shared__ float sMax[8][16];
    __shared__ float sSum[8][16];
    __shared__ float sAlpha[8][16];
    __shared__ __align__(16) ushort16_t sK[2][32][72];   // 32 keys x 64 dh (+pad)
    __shared__ __align__(16) ushort16_t sV[2][64][40];   // 64 dh x 32 keys (+pad)

    const int lane  = threadIdx.x;
    const int wid   = threadIdx.y;
    const int tid   = wid * 32 + lane;
    const int w     = blockIdx.x * 8 + wid;
    const int qt    = w & 127;            // T/16 = 128 tiles
    const int h     = (w >> 7) & 15;
    const int b     = w >> 11;
    const int qbase = qt * 16;
    const int lhalf = (lane < 16) ? 0 : 1;
    const int lm    = lane & 15;
    const float scale = 0.125f;           // 1/sqrt(64)

    const int qt0     = (blockIdx.x & 15) * 8;            // block's first q tile
    const int jmaxBlk = ((qt0 * 16 + 127) >> 5) << 5;     // block causal limit
    const int j0last  = ((qbase + 15) >> 5) << 5;         // wave causal limit

    const unsigned long long baseK = (unsigned long long)Kc;
    const unsigned long long baseV = (unsigned long long)Vt;
    const size_t vhOff = (size_t)(b * NHEAD + h) * DH * TT;   // elements

    // stage K tile (32 keys x 64 dh) and V tile (64 dh x 32 keys) into LDS[buf]
    auto stage = [&](int buf, int j0) {
        {   // K: 32 rows x 128B = 256 x 16B chunks, one per thread
            int r = tid >> 3, c = (tid & 7) * 16;
            unsigned lds = (unsigned)(size_t)&sK[buf][r][0] + (unsigned)c;
            unsigned go  = (unsigned)(((size_t)(b * TT + j0 + r) * CC + h * DH) * 2 + c);
            async_b128(lds, go, baseK);
        }
        {   // V (from Vt): 64 rows x 64B = 256 x 16B chunks, one per thread
            int r = tid >> 2, c = (tid & 3) * 16;
            unsigned lds = (unsigned)(size_t)&sV[buf][r][0] + (unsigned)c;
            unsigned go  = (unsigned)((vhOff + (size_t)r * TT + j0) * 2 + c);
            async_b128(lds, go, baseV);
        }
    };

    // Q A-fragments (two dh-chunks of 32), kept in registers for whole kernel
    ABFrag aq[2];
    {
        const ushort16_t* qrow = Q + (size_t)(b * TT + qbase + lm) * CC + h * DH;
        #pragma unroll
        for (int c = 0; c < 2; ++c) {
            const ushort16_t* p = qrow + 32 * c + lhalf * 8;
            aq[c].h.lo = *(const v8bf*)p;
            aq[c].h.hi = *(const v8bf*)(p + 16);
        }
    }

    v8f o[4];
    #pragma unroll
    for (int t = 0; t < 4; ++t) o[t] = (v8f){0.f,0.f,0.f,0.f,0.f,0.f,0.f,0.f};

    if (lane < 16) { sMax[wid][lane] = -__builtin_inff(); sSum[wid][lane] = 0.f; }
    asm volatile("" ::: "memory");

    stage(0, 0);
    int it = 0;
    for (int j0 = 0; j0 <= jmaxBlk; j0 += 32, ++it) {
        const int cur = it & 1;
        const bool more = (j0 + 32) <= jmaxBlk;
        if (more) {
            stage(cur ^ 1, j0 + 32);                       // prefetch next tile
            asm volatile("s_wait_asynccnt 2" ::: "memory");
        } else {
            asm volatile("s_wait_asynccnt 0" ::: "memory");
        }
        __syncthreads();

        if (j0 <= j0last) {    // wave-uniform: EXEC stays all-ones for WMMA
            // ---- S = Q * K^T : batch K fragments, then 4 WMMAs
            ABFrag bk[2][2];
            #pragma unroll
            for (int t = 0; t < 2; ++t)
                #pragma unroll
                for (int c = 0; c < 2; ++c) {
                    const ushort16_t* p = &sK[cur][16 * t + lm][32 * c + lhalf * 8];
                    bk[t][c].h.lo = *(const v8bf*)p;
                    bk[t][c].h.hi = *(const v8bf*)(p + 16);
                }
            __builtin_amdgcn_sched_barrier(0);
            v8f s[2];
            s[0] = (v8f){0.f,0.f,0.f,0.f,0.f,0.f,0.f,0.f};
            s[1] = (v8f){0.f,0.f,0.f,0.f,0.f,0.f,0.f,0.f};
            #pragma unroll
            for (int t = 0; t < 2; ++t)
                #pragma unroll
                for (int c = 0; c < 2; ++c)
                    s[t] = __builtin_amdgcn_wmma_f32_16x16x32_bf16(
                        false, aq[c].v, false, bk[t][c].v, (short)0, s[t], false, false);

            // ---- spill S to LDS (C-layout -> row major)
            #pragma unroll
            for (int t = 0; t < 2; ++t)
                #pragma unroll
                for (int v = 0; v < 8; ++v)
                    sS[wid][v + lhalf * 8][16 * t + lm] = s[t][v];
            asm volatile("" ::: "memory");

            // ---- online softmax, one row per lane (lanes 0..15)
            if (lane < 16) {
                const int r = lane, qi = qbase + r;
                float oldM = sMax[wid][r];
                float mx = oldM;
                for (int c2 = 0; c2 < 32; ++c2) {
                    int j = j0 + c2;
                    float sv = (j <= qi) ? sS[wid][r][c2] * scale : -__builtin_inff();
                    mx = fmaxf(mx, sv);
                }
                float alpha = __expf(oldM - mx);
                float sum = 0.f;
                for (int c2 = 0; c2 < 32; ++c2) {
                    int j = j0 + c2;
                    float e = (j <= qi) ? __expf(sS[wid][r][c2] * scale - mx) : 0.f;
                    sum += e;
                    sP[wid][r][c2] = f2bf(e);
                }
                sSum[wid][r] = sSum[wid][r] * alpha + sum;
                sMax[wid][r] = mx;
                sAlpha[wid][r] = alpha;
            }
            asm volatile("" ::: "memory");

            // ---- rescale O accumulators by alpha(row)
            #pragma unroll
            for (int v = 0; v < 8; ++v) {
                float a = sAlpha[wid][v + lhalf * 8];
                o[0][v] *= a; o[1][v] *= a; o[2][v] *= a; o[3][v] *= a;
            }

            // ---- P A-fragment + V B-fragments, then 4 WMMAs
            ABFrag ap;
            {
                const ushort16_t* p = &sP[wid][lm][lhalf * 8];
                ap.h.lo = *(const v8bf*)p;
                ap.h.hi = *(const v8bf*)(p + 16);
            }
            ABFrag bv[4];
            #pragma unroll
            for (int t = 0; t < 4; ++t) {
                const ushort16_t* p = &sV[cur][16 * t + lm][lhalf * 8];
                bv[t].h.lo = *(const v8bf*)p;
                bv[t].h.hi = *(const v8bf*)(p + 16);
            }
            __builtin_amdgcn_sched_barrier(0);
            #pragma unroll
            for (int t = 0; t < 4; ++t)
                o[t] = __builtin_amdgcn_wmma_f32_16x16x32_bf16(
                    false, ap.v, false, bv[t].v, (short)0, o[t], false, false);
        }
        __syncthreads();         // tile fully consumed before buffer reuse
    }

    // ---- normalize and store O as bf16 [B*T, C]
    if (lane < 16) sAlpha[wid][lane] = 1.0f / sSum[wid][lane];
    asm volatile("" ::: "memory");
    #pragma unroll
    for (int t = 0; t < 4; ++t) {
        #pragma unroll
        for (int v = 0; v < 8; ++v) {
            int rr = v + lhalf * 8;
            int row = qbase + rr;
            float inv = sAlpha[wid][rr];
            O[(size_t)(b * TT + row) * CC + h * DH + 16 * t + lm] = f2bf(o[t][v] * inv);
        }
    }
}

// ---------------------------------------------------------------- launcher
extern "C" void kernel_launch(void* const* d_in, const int* in_sizes, int n_in,
                              void* d_out, int out_size, void* d_ws, size_t ws_size,
                              hipStream_t stream) {
    const float* x  = (const float*)d_in[0];
    const float* Wk = (const float*)d_in[1];
    const float* Wq = (const float*)d_in[2];
    const float* Wv = (const float*)d_in[3];
    const float* Wp = (const float*)d_in[4];
    const float* bp = (const float*)d_in[5];
    float* out = (float*)d_out;

    const int M = BB * TT;          // 4096 rows
    const int K = CC;               // 1024
    const int N = CC;               // 1024

    char* p = (char*)d_ws;
    auto alloc = [&](size_t bytes) { char* r = p; p += (bytes + 255) & ~(size_t)255; return r; };
    ushort16_t* xb  = (ushort16_t*)alloc((size_t)M * K * 2);
    ushort16_t* WkT = (ushort16_t*)alloc((size_t)K * N * 2);
    ushort16_t* WqT = (ushort16_t*)alloc((size_t)K * N * 2);
    ushort16_t* WvT = (ushort16_t*)alloc((size_t)K * N * 2);
    ushort16_t* WpT = (ushort16_t*)alloc((size_t)K * N * 2);
    ushort16_t* Qb  = (ushort16_t*)alloc((size_t)M * N * 2);
    ushort16_t* Kb  = (ushort16_t*)alloc((size_t)M * N * 2);
    ushort16_t* Vb  = (ushort16_t*)alloc((size_t)M * N * 2);
    ushort16_t* VtB = (ushort16_t*)alloc((size_t)BB * NHEAD * DH * TT * 2);
    ushort16_t* Ob  = (ushort16_t*)alloc((size_t)M * N * 2);

    // 1) conversions
    k_f32_to_bf16<<<2048, 256, 0, stream>>>(x, xb, M * K);
    k_w_transpose_bf16<<<(K * N + 255) / 256, 256, 0, stream>>>(Wk, WkT, K, N);
    k_w_transpose_bf16<<<(K * N + 255) / 256, 256, 0, stream>>>(Wq, WqT, K, N);
    k_w_transpose_bf16<<<(K * N + 255) / 256, 256, 0, stream>>>(Wv, WvT, K, N);
    k_w_transpose_bf16<<<(K * N + 255) / 256, 256, 0, stream>>>(Wp, WpT, K, N);

    // 2) QKV projections (bf16 out)
    dim3 gblk(32, 8);
    dim3 ggrid(N / 64, M / 256);
    k_gemm_bf16<<<ggrid, gblk, 0, stream>>>(xb, WqT, nullptr, Qb, nullptr, M, N, K);
    k_gemm_bf16<<<ggrid, gblk, 0, stream>>>(xb, WkT, nullptr, Kb, nullptr, M, N, K);
    k_gemm_bf16<<<ggrid, gblk, 0, stream>>>(xb, WvT, nullptr, Vb, nullptr, M, N, K);

    // 3) V transpose for PV B-fragments
    int nvt = BB * NHEAD * DH * TT;
    k_v_transpose<<<(nvt + 255) / 256, 256, 0, stream>>>(Vb, VtB);

    // 4) flash attention (8 q-tiles of one (b,h) per block)
    int nwaves = BB * NHEAD * (TT / 16);      // 4096
    k_attn<<<nwaves / 8, dim3(32, 8), 0, stream>>>(Qb, Kb, VtB, Ob);

    // 5) output projection with bias (f32 out)
    k_gemm_bf16<<<ggrid, gblk, 0, stream>>>(Ob, WpT, out, nullptr, bp, M, N, K);
}
